// Transformer_90598040141850
// MI455X (gfx1250) — matrix operand order repaired and checked by reading, not compiled, not run on hardware
//
#include <hip/hip_runtime.h>
#include <hip/hip_bf16.h>
#include <cstdint>

typedef __bf16 bf16;
typedef __attribute__((ext_vector_type(16))) __bf16 v16bf;
typedef __attribute__((ext_vector_type(8)))  float  v8f;

#define DEV static __device__ __forceinline__

// ---------------------------------------------------------------------------
// Async global->LDS copy (CDNA5 Tensor/async path, tracked by ASYNCcnt).
// One instruction moves 16B per lane. ldsoff = LDS byte address (low 32 bits
// of the generic shared pointer), g = per-lane global address.
// ---------------------------------------------------------------------------
DEV void async_b128(uint32_t ldsoff, const bf16* g) {
  asm volatile("global_load_async_to_lds_b128 %0, %1, off"
               :: "v"(ldsoff), "v"(g) : "memory");
}
DEV void wait_async0() {
  asm volatile("s_wait_asynccnt 0x0" ::: "memory");
}

// ---------------------------------------------------------------------------
// WMMA fragment extraction from an LDS row pointer (branch-free).
// A 16x32 layout (ISA 7.12.2): f[0..7] = row[half*8+0..7],
//                              f[8..15] = row[16+half*8+0..7]
// B 32x16 layout (same k order): f[0..7] = row[half*8+0..7], f[8..15] = row[16+half*8+0..7]
// Each compiles to 2x ds_load_b128.
// ---------------------------------------------------------------------------
DEV v16bf frag_a(const bf16* __restrict__ row, int half) {
  const bf16* p0 = row + half * 8;
  const bf16* p1 = row + 16 + half * 8;
  v16bf f;
#pragma unroll
  for (int e = 0; e < 8; ++e) { f[e] = p0[e]; f[8 + e] = p1[e]; }
  return f;
}
DEV v16bf frag_b(const bf16* __restrict__ row, int half) {
  const bf16* p0 = row + half * 8;
  const bf16* p1 = row + 16 + half * 8;
  v16bf f;
#pragma unroll
  for (int e = 0; e < 8; ++e) { f[e] = p0[e]; f[8 + e] = p1[e]; }
  return f;
}

// ---------------------------------------------------------------------------
// Generic batched GEMM: C = A * B with B given transposed (N,K) row-major.
// bf16 operands, f32 WMMA accumulate. Block = 256 threads = 8 waves (4x2),
// wave tile 32x64 (2x4 WMMA tiles) -> block tile 128x128, K-step 32.
// A/B slabs (128x32 each) are staged in LDS via async-load-to-LDS with
// double buffering; fragments are read with ds_load_b128. Rows are padded
// to 80B (20 banks) for conflict-free 16-row fragment reads.
// Out-of-range rows/cols are CLAMPED on load (results never stored).
// Batch z decomposed (z>>2, z&3) with stride pairs for (batch,head) views.
// ---------------------------------------------------------------------------
template<bool OUT_BF16>
__global__ __launch_bounds__(256)
void gemm_wmma_k(const bf16* __restrict__ A, int lda, long sAo, long sAi,
                 const bf16* __restrict__ Bt, int ldb, long sBo, long sBi,
                 void* __restrict__ Cout, int ldc, long sCo, long sCi,
                 int M, int N, int K, const float* __restrict__ bias)
{
  __shared__ bf16 smem[2][2][128 * 40];   // [buf][A/B][row*40+e], 40KB
  const uint32_t SLABB = 128 * 40 * 2;    // bytes per A or B slab
  const uint32_t BUFB  = 2 * SLABB;       // bytes per buffer

  int tid  = threadIdx.x;
  int lane = tid & 31;
  int wave = tid >> 5;
  int z    = blockIdx.z;
  A  += (long)(z >> 2) * sAo + (long)(z & 3) * sAi;
  Bt += (long)(z >> 2) * sBo + (long)(z & 3) * sBi;
  long co = (long)(z >> 2) * sCo + (long)(z & 3) * sCi;

  // ---- cooperative slab loader: each thread owns 2 A-chunks + 2 B-chunks
  int part  = tid & 3;            // 16B chunk within a 64B row
  int rowc0 = tid >> 2;           // rows 0..63
  int rowc1 = 64 + rowc0;         // rows 64..127
  int mg0 = blockIdx.x * 128 + rowc0; if (mg0 > M - 1) mg0 = M - 1;
  int mg1 = blockIdx.x * 128 + rowc1; if (mg1 > M - 1) mg1 = M - 1;
  int ng0 = blockIdx.y * 128 + rowc0; if (ng0 > N - 1) ng0 = N - 1;
  int ng1 = blockIdx.y * 128 + rowc1; if (ng1 > N - 1) ng1 = N - 1;
  const bf16* gA0 = A  + (size_t)mg0 * lda + part * 8;
  const bf16* gA1 = A  + (size_t)mg1 * lda + part * 8;
  const bf16* gB0 = Bt + (size_t)ng0 * ldb + part * 8;
  const bf16* gB1 = Bt + (size_t)ng1 * ldb + part * 8;
  uint32_t sbase = (uint32_t)(uintptr_t)(&smem[0][0][0]);
  uint32_t offA0 = sbase + rowc0 * 80 + part * 16;
  uint32_t offA1 = sbase + rowc1 * 80 + part * 16;
  uint32_t offB0 = offA0 + SLABB;
  uint32_t offB1 = offA1 + SLABB;

  auto issue = [&](int buf, int k) {
    uint32_t bo = (uint32_t)buf * BUFB;
    async_b128(offA0 + bo, gA0 + k);
    async_b128(offA1 + bo, gA1 + k);
    async_b128(offB0 + bo, gB0 + k);
    async_b128(offB1 + bo, gB1 + k);
  };

  // ---- per-wave compute tile
  int wm = wave & 3, wn = wave >> 2;
  int lm = lane & 15, half = lane >> 4;
  int m0 = blockIdx.x * 128 + wm * 32;
  int n0 = blockIdx.y * 128 + wn * 64;

  v8f acc[2][4];
#pragma unroll
  for (int i = 0; i < 2; ++i)
#pragma unroll
    for (int t = 0; t < 4; ++t)
#pragma unroll
      for (int v = 0; v < 8; ++v) acc[i][t][v] = 0.0f;

  // ---- double-buffered pipeline
  issue(0, 0);
  wait_async0();
  __syncthreads();

  int nk = K >> 5;
  for (int ks = 0; ks < nk; ++ks) {
    int k   = ks << 5;
    int buf = ks & 1;
    if (ks + 1 < nk) {
      __builtin_prefetch(gA0 + k + 96, 0, 1);   // L2 prefetch 2 slabs ahead
      issue(buf ^ 1, k + 32);
    }

    const bf16* a0r = &smem[buf][0][(wm * 32 + lm) * 40];
    const bf16* a1r = &smem[buf][0][(wm * 32 + 16 + lm) * 40];
    v16bf a0 = frag_a(a0r, half);
    v16bf a1 = frag_a(a1r, half);
    v16bf b0 = frag_b(&smem[buf][1][(wn * 64 +  0 + lm) * 40], half);
    v16bf b1 = frag_b(&smem[buf][1][(wn * 64 + 16 + lm) * 40], half);
    v16bf b2 = frag_b(&smem[buf][1][(wn * 64 + 32 + lm) * 40], half);
    v16bf b3 = frag_b(&smem[buf][1][(wn * 64 + 48 + lm) * 40], half);

    acc[0][0] = __builtin_amdgcn_wmma_f32_16x16x32_bf16(false, a0, false, b0, (short)0, acc[0][0], false, false);
    acc[0][1] = __builtin_amdgcn_wmma_f32_16x16x32_bf16(false, a0, false, b1, (short)0, acc[0][1], false, false);
    acc[0][2] = __builtin_amdgcn_wmma_f32_16x16x32_bf16(false, a0, false, b2, (short)0, acc[0][2], false, false);
    acc[0][3] = __builtin_amdgcn_wmma_f32_16x16x32_bf16(false, a0, false, b3, (short)0, acc[0][3], false, false);
    acc[1][0] = __builtin_amdgcn_wmma_f32_16x16x32_bf16(false, a1, false, b0, (short)0, acc[1][0], false, false);
    acc[1][1] = __builtin_amdgcn_wmma_f32_16x16x32_bf16(false, a1, false, b1, (short)0, acc[1][1], false, false);
    acc[1][2] = __builtin_amdgcn_wmma_f32_16x16x32_bf16(false, a1, false, b2, (short)0, acc[1][2], false, false);
    acc[1][3] = __builtin_amdgcn_wmma_f32_16x16x32_bf16(false, a1, false, b3, (short)0, acc[1][3], false, false);

    if (ks + 1 < nk) {
      wait_async0();      // own async copies into buf^1 have landed
      __syncthreads();    // everyone's copies have landed / reads of buf done
    }
  }

  // ---- store
  float* Cf = (float*)Cout;
  bf16*  Cb = (bf16*)Cout;
#pragma unroll
  for (int ti = 0; ti < 2; ++ti)
#pragma unroll
    for (int tj = 0; tj < 4; ++tj)
#pragma unroll
      for (int v = 0; v < 8; ++v) {
        int m = m0 + ti * 16 + v + half * 8;
        int n = n0 + tj * 16 + lm;
        if (m < M && n < N) {
          float val = acc[ti][tj][v];
          if (bias) val += bias[n];
          long idx = co + (long)m * ldc + n;
          if (OUT_BF16) Cb[idx] = (bf16)val; else Cf[idx] = val;
        }
      }
}

// ---------------------------------------------------------------------------
// Elementwise / reduction kernels
// ---------------------------------------------------------------------------
__global__ void cvt_bf16_k(const float* __restrict__ in, bf16* __restrict__ out, long n) {
  long i = (long)blockIdx.x * 256 + threadIdx.x;
  if (i < n) out[i] = (bf16)in[i];
}

// im2col for stride==kernel (non-overlapping patches): dst (M,K) bf16.
// src: (B=8, 4096, 512) f32. All dims powers of two -> shift/mask only.
__global__ void pack_patches_k(const float* __restrict__ src, bf16* __restrict__ dst,
                               int srLog, int PLog, int KLog, long total)
{
  long t = (long)blockIdx.x * 256 + threadIdx.x;
  if (t >= total) return;
  int  k = (int)(t & ((1 << KLog) - 1));
  long m = t >> KLog;
  int ssLog = 2 * srLog;
  int c  = k >> ssLog;
  int rj = k & ((1 << ssLog) - 1);
  int i  = rj >> srLog;
  int j  = rj & ((1 << srLog) - 1);
  int px = (int)(m & ((1 << PLog) - 1));
  int py = (int)((m >> PLog) & ((1 << PLog) - 1));
  int b  = (int)(m >> (2 * PLog));
  int y  = (py << srLog) + i, x = (px << srLog) + j;
  dst[t] = (bf16)src[(((long)(b << 12) + (y << 6) + x) << 9) + c];
}

DEV float wred_sum(float v) {
#pragma unroll
  for (int o = 16; o > 0; o >>= 1) v += __shfl_xor(v, o, 32);
  return v;
}
DEV float wred_max(float v) {
#pragma unroll
  for (int o = 16; o > 0; o >>= 1) v = fmaxf(v, __shfl_xor(v, o, 32));
  return v;
}

// conv f32 (rows x 512) + bias -> LayerNorm(g,b) -> relu6 -> bf16
__global__ __launch_bounds__(256)
void ln_relu6_k(const float* __restrict__ tin, const float* __restrict__ cb,
                const float* __restrict__ g, const float* __restrict__ bb,
                bf16* __restrict__ out)
{
  long r = blockIdx.x;
  int  t = threadIdx.x;
  float x0 = tin[r * 512 + t]       + cb[t];
  float x1 = tin[r * 512 + 256 + t] + cb[256 + t];
  float s = x0 + x1, sq = x0 * x0 + x1 * x1;
  __shared__ float sh1[8], sh2[8];
  s = wred_sum(s); sq = wred_sum(sq);
  int lane = t & 31, w = t >> 5;
  if (lane == 0) { sh1[w] = s; sh2[w] = sq; }
  __syncthreads();
  if (t < 32) {
    float a  = (lane < 8) ? sh1[lane] : 0.f;
    float b2 = (lane < 8) ? sh2[lane] : 0.f;
    a = wred_sum(a); b2 = wred_sum(b2);
    if (lane == 0) { sh1[0] = a; sh2[0] = b2; }
  }
  __syncthreads();
  float mean = sh1[0] * (1.0f / 512.0f);
  float var  = sh2[0] * (1.0f / 512.0f) - mean * mean;
  float inv  = rsqrtf(var + 1e-5f);
  float y0 = (x0 - mean) * inv * g[t]       + bb[t];
  float y1 = (x1 - mean) * inv * g[256 + t] + bb[256 + t];
  y0 = fminf(fmaxf(y0, 0.f), 6.f);
  y1 = fminf(fmaxf(y1, 0.f), 6.f);
  out[r * 512 + t]       = (bf16)y0;
  out[r * 512 + 256 + t] = (bf16)y1;
}

// 3x3 depthwise conv (pad 1) on v-image + bias, plus residual v -> bf16,
// written TRANSPOSED per (b,h): vs[((b*4+h)*64 + d)*L + l]  ((N,K) for AV GEMM)
__global__ void dwconv_vs_k(const float* __restrict__ vhf, const float* __restrict__ lw,
                            const float* __restrict__ lb, bf16* __restrict__ vs,
                            int wwLog)
{
  int  ww = 1 << wwLog;
  int  LLog = 2 * wwLog;
  int  L = 1 << LLog;
  long total = (long)8 * 4 * 64 * L;
  long t = (long)blockIdx.x * 256 + threadIdx.x;
  if (t >= total) return;
  int l = (int)(t & (L - 1));
  long r = t >> LLog;
  int d = (int)(r & 63);
  int h = (int)((r >> 6) & 3);
  int b = (int)(r >> 8);
  int c = h * 64 + d;
  int y = l >> wwLog, x = l & (ww - 1);
  float acc = lb[c];
#pragma unroll
  for (int dy = -1; dy <= 1; ++dy)
#pragma unroll
    for (int dx = -1; dx <= 1; ++dx) {
      int yy = y + dy, xx = x + dx;
      if (yy >= 0 && yy < ww && xx >= 0 && xx < ww)
        acc += lw[c * 9 + (dy + 1) * 3 + (dx + 1)] *
               vhf[(((long)b << LLog) + (yy << wwLog) + xx) * 256 + c];
    }
  float vh = vhf[(((long)b << LLog) + l) * 256 + c];
  vs[t] = (bf16)(vh + acc);
}

// scaled softmax along L: f32 logits -> bf16 attn (rows = 8*4*200)
__global__ __launch_bounds__(256)
void softmax_k(const float* __restrict__ logits, bf16* __restrict__ attn, int L)
{
  long row = blockIdx.x;
  const float* in = logits + row * (long)L;
  bf16* out = attn + row * (long)L;
  int t = threadIdx.x, lane = t & 31, w = t >> 5;
  __shared__ float shm[8], shs[8];
  float mx = -3.0e38f;
  for (int i = t; i < L; i += 256) mx = fmaxf(mx, in[i] * 0.125f);
  mx = wred_max(mx);
  if (lane == 0) shm[w] = mx;
  __syncthreads();
  if (t < 32) {
    float a = (lane < 8) ? shm[lane] : -3.0e38f;
    a = wred_max(a);
    if (lane == 0) shm[0] = a;
  }
  __syncthreads();
  mx = shm[0];
  float s = 0.f;
  for (int i = t; i < L; i += 256) s += __expf(in[i] * 0.125f - mx);
  s = wred_sum(s);
  if (lane == 0) shs[w] = s;
  __syncthreads();
  if (t < 32) {
    float a = (lane < 8) ? shs[lane] : 0.f;
    a = wred_sum(a);
    if (lane == 0) shs[0] = a;
  }
  __syncthreads();
  float inv = 1.0f / shs[0];
  for (int i = t; i < L; i += 256) out[i] = (bf16)(__expf(in[i] * 0.125f - mx) * inv);
}

// Replicate the reference's raw reshape: xh (B,4,200,64) ->
// flatten of (B,200,4,64) reinterpreted as (Q,B,256), concat halves -> (Q,B,512)
__global__ void remap_x_k(const float* __restrict__ xh, bf16* __restrict__ xout, int br)
{
  long f = (long)blockIdx.x * 256 + threadIdx.x;
  if (f >= 409600) return;
  int d = (int)(f & 63);
  long r = f >> 6;
  int h = (int)(r & 3); r >>= 2;
  int qq = (int)(r % 200);
  int b  = (int)(r / 200);
  float v = xh[(((long)b * 4 + h) * 200 + qq) * 64 + d];
  long qi = f / 2048;
  long bi = (f >> 8) & 7;
  long c  = f & 255;
  xout[qi * 4096 + bi * 512 + br * 256 + c] = (bf16)v;
}

// ---------------------------------------------------------------------------
// Host orchestration
// ---------------------------------------------------------------------------
extern "C" void kernel_launch(void* const* d_in, const int* in_sizes, int n_in,
                              void* d_out, int out_size, void* d_ws, size_t ws_size,
                              hipStream_t stream)
{
  (void)in_sizes; (void)n_in; (void)out_size; (void)ws_size;
  const float* query  = (const float*)d_in[0];
  const float* key    = (const float*)d_in[1];
  const float* value  = (const float*)d_in[2];
  const float* q_w    = (const float*)d_in[3];
  const float* s11w   = (const float*)d_in[4];
  const float* s11b   = (const float*)d_in[5];
  const float* n11g   = (const float*)d_in[6];
  const float* n11b   = (const float*)d_in[7];
  const float* s12w   = (const float*)d_in[8];
  const float* s12b   = (const float*)d_in[9];
  const float* n12g   = (const float*)d_in[10];
  const float* n12b   = (const float*)d_in[11];
  const float* s21w   = (const float*)d_in[12];
  const float* s21b   = (const float*)d_in[13];
  const float* n21g   = (const float*)d_in[14];
  const float* n21b   = (const float*)d_in[15];
  const float* s22w   = (const float*)d_in[16];
  const float* s22b   = (const float*)d_in[17];
  const float* n22g   = (const float*)d_in[18];
  const float* n22b   = (const float*)d_in[19];
  const float* k1_w   = (const float*)d_in[20];
  const float* v1_w   = (const float*)d_in[21];
  const float* k2_w   = (const float*)d_in[22];
  const float* v2_w   = (const float*)d_in[23];
  const float* lc1w   = (const float*)d_in[24];
  const float* lc1b   = (const float*)d_in[25];
  const float* lc2w   = (const float*)d_in[26];
  const float* lc2b   = (const float*)d_in[27];
  const float* projw  = (const float*)d_in[28];
  const float* projb  = (const float*)d_in[29];
  float* out = (float*)d_out;

  // workspace carve-up (256B aligned)
  char* base = (char*)d_ws;
  size_t off = 0;
  auto alloc = [&](size_t bytes) -> void* {
    void* p = base + off;
    off = (off + bytes + 255) & ~(size_t)255;
    return p;
  };
  bf16* QA    = (bf16*)alloc(819200 * 2);
  bf16* WQ    = (bf16*)alloc(262144 * 2);
  bf16* WS11  = (bf16*)alloc(4194304 * 2);
  bf16* WS12  = (bf16*)alloc(4194304 * 2);
  bf16* WS21  = (bf16*)alloc(1048576 * 2);
  bf16* WS22  = (bf16*)alloc(1048576 * 2);
  bf16* WK1   = (bf16*)alloc(131072 * 2);
  bf16* WV1   = (bf16*)alloc(131072 * 2);
  bf16* WK2   = (bf16*)alloc(131072 * 2);
  bf16* WV2   = (bf16*)alloc(131072 * 2);
  bf16* WP    = (bf16*)alloc(262144 * 2);
  bf16* QB    = (bf16*)alloc(819200 * 2);
  bf16* PATCH = (bf16*)alloc((size_t)16777216 * 2);
  float* CONVF= (float*)alloc((size_t)4194304 * 4);
  bf16* KEY1  = (bf16*)alloc(1048576 * 2);
  bf16* VAL1  = (bf16*)alloc(1048576 * 2);
  bf16* KEY2  = (bf16*)alloc(4194304 * 2);
  bf16* VAL2  = (bf16*)alloc(4194304 * 2);
  bf16* K1    = (bf16*)alloc(524288 * 2);
  bf16* K2    = (bf16*)alloc(2097152 * 2);
  float* V1F  = (float*)alloc(524288 * 4);
  float* V2F  = (float*)alloc(2097152 * 4);
  bf16* VS1   = (bf16*)alloc(524288 * 2);
  bf16* VS2   = (bf16*)alloc(2097152 * 2);
  float* LOG  = (float*)alloc((size_t)6553600 * 4);
  bf16* ATT   = (bf16*)alloc((size_t)6553600 * 2);
  float* XH1  = (float*)alloc(409600 * 4);
  float* XH2  = (float*)alloc(409600 * 4);
  bf16* XB    = (bf16*)alloc(819200 * 2);

  auto cvt = [&](const float* s, bf16* d, long n) {
    cvt_bf16_k<<<dim3((unsigned)((n + 255) / 256)), dim3(256), 0, stream>>>(s, d, n);
  };

  cvt(query, QA, 819200);
  cvt(q_w,   WQ, 262144);
  cvt(s11w,  WS11, 4194304);
  cvt(s12w,  WS12, 4194304);
  cvt(s21w,  WS21, 1048576);
  cvt(s22w,  WS22, 1048576);
  cvt(k1_w,  WK1, 131072);
  cvt(v1_w,  WV1, 131072);
  cvt(k2_w,  WK2, 131072);
  cvt(v2_w,  WV2, 131072);
  cvt(projw, WP,  262144);

  gemm_wmma_k<true><<<dim3(13, 4, 1), 256, 0, stream>>>(
      QA, 512, 0, 0, WQ, 512, 0, 0, QB, 512, 0, 0, 1600, 512, 512, nullptr);

  pack_patches_k<<<dim3(65536), 256, 0, stream>>>(key, PATCH, 2, 4, 13, (long)2048 * 8192);
  gemm_wmma_k<false><<<dim3(16, 4, 1), 256, 0, stream>>>(
      PATCH, 8192, 0, 0, WS11, 8192, 0, 0, CONVF, 512, 0, 0, 2048, 512, 8192, nullptr);
  ln_relu6_k<<<dim3(2048), 256, 0, stream>>>(CONVF, s11b, n11g, n11b, KEY1);

  pack_patches_k<<<dim3(65536), 256, 0, stream>>>(value, PATCH, 2, 4, 13, (long)2048 * 8192);
  gemm_wmma_k<false><<<dim3(16, 4, 1), 256, 0, stream>>>(
      PATCH, 8192, 0, 0, WS12, 8192, 0, 0, CONVF, 512, 0, 0, 2048, 512, 8192, nullptr);
  ln_relu6_k<<<dim3(2048), 256, 0, stream>>>(CONVF, s12b, n12g, n12b, VAL1);

  pack_patches_k<<<dim3(65536), 256, 0, stream>>>(key, PATCH, 1, 5, 11, (long)8192 * 2048);
  gemm_wmma_k<false><<<dim3(64, 4, 1), 256, 0, stream>>>(
      PATCH, 2048, 0, 0, WS21, 2048, 0, 0, CONVF, 512, 0, 0, 8192, 512, 2048, nullptr);
  ln_relu6_k<<<dim3(8192), 256, 0, stream>>>(CONVF, s21b, n21g, n21b, KEY2);

  pack_patches_k<<<dim3(65536), 256, 0, stream>>>(value, PATCH, 1, 5, 11, (long)8192 * 2048);
  gemm_wmma_k<false><<<dim3(64, 4, 1), 256, 0, stream>>>(
      PATCH, 2048, 0, 0, WS22, 2048, 0, 0, CONVF, 512, 0, 0, 8192, 512, 2048, nullptr);
  ln_relu6_k<<<dim3(8192), 256, 0, stream>>>(CONVF, s22b, n22g, n22b, VAL2);

  gemm_wmma_k<true><<<dim3(16, 2, 1), 256, 0, stream>>>(
      KEY1, 512, 0, 0, WK1, 512, 0, 0, K1, 256, 0, 0, 2048, 256, 512, nullptr);
  gemm_wmma_k<false><<<dim3(16, 2, 1), 256, 0, stream>>>(
      VAL1, 512, 0, 0, WV1, 512, 0, 0, V1F, 256, 0, 0, 2048, 256, 512, nullptr);
  gemm_wmma_k<true><<<dim3(64, 2, 1), 256, 0, stream>>>(
      KEY2, 512, 0, 0, WK2, 512, 0, 0, K2, 256, 0, 0, 8192, 256, 512, nullptr);
  gemm_wmma_k<false><<<dim3(64, 2, 1), 256, 0, stream>>>(
      VAL2, 512, 0, 0, WV2, 512, 0, 0, V2F, 256, 0, 0, 8192, 256, 512, nullptr);

  dwconv_vs_k<<<dim3(2048), 256, 0, stream>>>(V1F, lc1w, lc1b, VS1, 4);
  dwconv_vs_k<<<dim3(8192), 256, 0, stream>>>(V2F, lc2w, lc2b, VS2, 5);

  gemm_wmma_k<false><<<dim3(2, 2, 32), 256, 0, stream>>>(
      QB, 512, 102400, 64,
      K1, 256, 65536, 64,
      LOG, 256, 204800, 51200,
      200, 256, 64, nullptr);
  softmax_k<<<dim3(6400), 256, 0, stream>>>(LOG, ATT, 256);
  gemm_wmma_k<false><<<dim3(2, 1, 32), 256, 0, stream>>>(
      ATT, 256, 204800, 51200,
      VS1, 256, 65536, 16384,
      XH1, 64, 51200, 12800,
      200, 64, 256, nullptr);
  remap_x_k<<<dim3(1600), 256, 0, stream>>>(XH1, XB, 0);

  gemm_wmma_k<false><<<dim3(2, 8, 32), 256, 0, stream>>>(
      QB + 256, 512, 102400, 64,
      K2, 256, 262144, 64,
      LOG, 1024, 819200, 204800,
      200, 1024, 64, nullptr);
  softmax_k<<<dim3(6400), 256, 0, stream>>>(LOG, ATT, 1024);
  gemm_wmma_k<false><<<dim3(2, 1, 32), 256, 0, stream>>>(
      ATT, 1024, 819200, 204800,
      VS2, 1024, 262144, 65536,
      XH2, 64, 51200, 12800,
      200, 64, 1024, nullptr);
  remap_x_k<<<dim3(1600), 256, 0, stream>>>(XH2, XB, 1);

  gemm_wmma_k<false><<<dim3(13, 4, 1), 256, 0, stream>>>(
      XB, 512, 0, 0, WP, 512, 0, 0, out, 512, 0, 0, 1600, 512, 512, projb);
}
